// Diff_SSM_Block_68822555951849
// MI455X (gfx1250) — matrix operand order repaired and checked by reading, not compiled, not run on hardware
//
#include <hip/hip_runtime.h>
#include <hip/hip_bf16.h>
#include <math.h>

// ---- problem constants (match reference) ----
#define BB   4
#define LL   4096
#define HH   1024
#define DI   2048
#define BL   (BB*LL)        // 16384 tokens
#define N2   (2*DI)         // 4096
#define NS   16             // SSM state dim
#define DRr  64
#define XD   96             // DR + 2*NS
#define KC   4              // conv taps

typedef __attribute__((ext_vector_type(16))) __bf16 v16bf;
typedef __attribute__((ext_vector_type(8)))  float  v8f;

union Frag { uint4 q[2]; v16bf v; };

__device__ __forceinline__ unsigned short f2bf(float f) {
  union { float f; unsigned u; } v; v.f = f;
  unsigned u = v.u;
  unsigned r = (u + 0x7FFFu + ((u >> 16) & 1u)) >> 16;   // RNE
  return (unsigned short)r;
}
__device__ __forceinline__ float bf2f(unsigned short h) {
  union { unsigned u; float f; } v; v.u = ((unsigned)h) << 16;
  return v.f;
}
__device__ __forceinline__ float silu_f(float v) { return v / (1.f + __expf(-v)); }

// ---------------- fp32 -> bf16 conversion (weights) ----------------
__global__ void cvt_bf16_kernel(const float* __restrict__ in,
                                unsigned short* __restrict__ out, long long n) {
  long long i = (long long)blockIdx.x * blockDim.x + threadIdx.x;
  long long stride = (long long)gridDim.x * blockDim.x;
  for (; i < n; i += stride) out[i] = f2bf(in[i]);
}

// ---------------- LayerNorm over H=1024, one token per block ----------------
__global__ __launch_bounds__(256) void ln_kernel(const float* __restrict__ x,
                                                 unsigned short* __restrict__ outH) {
  __shared__ float s1[256], s2[256];
  int tid = threadIdx.x;
  size_t base = (size_t)blockIdx.x * HH;
  float v[4]; float s = 0.f;
#pragma unroll
  for (int j = 0; j < 4; ++j) { v[j] = x[base + tid + j * 256]; s += v[j]; }
  s1[tid] = s; __syncthreads();
  for (int off = 128; off > 0; off >>= 1) { if (tid < off) s1[tid] += s1[tid + off]; __syncthreads(); }
  float mu = s1[0] * (1.f / HH);
  float q = 0.f;
#pragma unroll
  for (int j = 0; j < 4; ++j) { float d = v[j] - mu; q += d * d; }
  s2[tid] = q; __syncthreads();
  for (int off = 128; off > 0; off >>= 1) { if (tid < off) s2[tid] += s2[tid + off]; __syncthreads(); }
  float rstd = rsqrtf(s2[0] * (1.f / HH) + 1e-6f);
#pragma unroll
  for (int j = 0; j < 4; ++j) outH[base + tid + j * 256] = f2bf((v[j] - mu) * rstd);
}

// ---------------- generic bf16 WMMA GEMM: C = act(A * W^T + bias) [*mul] [+res] ----------------
// A: M x K bf16 (lda), W: N x K bf16 (ldw).
// Wave tile 32(M) x 64(N): 2 A-frags x 4 B-frags -> 8 WMMAs per 32-K step.
// Block: 8 waves (4 in M, 2 in N) = 128 x 128 tile.
__global__ __launch_bounds__(256, 2)
void gemm_bf16_wmma(const unsigned short* __restrict__ A, int lda,
                    const unsigned short* __restrict__ W, int ldw,
                    const float* __restrict__ bias,
                    const float* __restrict__ mul,
                    const float* __restrict__ res,
                    float* __restrict__ outF,
                    unsigned short* __restrict__ outH,
                    int M, int Nn, int Kk, int ldc, int act) {
  int tid  = threadIdx.x;
  int wave = tid >> 5, lane = tid & 31;
  int lr = lane & 15, half = lane >> 4;
  int mwave = wave & 3, nwave = wave >> 2;
  int row0    = blockIdx.y * 128 + mwave * 32;
  int colBase = blockIdx.x * 128 + nwave * 64;

  v8f z{};
  v8f acc[2][4] = { { z, z, z, z }, { z, z, z, z } };

  const unsigned short* Arow0 = A + (size_t)(row0 + lr) * lda;
  const unsigned short* Arow1 = A + (size_t)(row0 + 16 + lr) * lda;
  // Hoist clamped W row pointers for the 4 column tiles out of the K-loop.
  const unsigned short* Wp[4];
#pragma unroll
  for (int t = 0; t < 4; ++t) {
    int bcol = colBase + t * 16 + lr;
    int bc = (bcol < Nn) ? bcol : 0;           // clamp keeps EXEC full; OOB cols never stored
    Wp[t] = W + (size_t)bc * ldw;
  }

  for (int kk = 0; kk < Kk; kk += 32) {
    int kb = kk + half * 8;                    // A frag: K 0..7 / 8..15 by lane-half, +16 upper
    Frag a0, a1;
    a0.q[0] = *(const uint4*)(Arow0 + kb);
    a0.q[1] = *(const uint4*)(Arow0 + kb + 16);
    a1.q[0] = *(const uint4*)(Arow1 + kb);
    a1.q[1] = *(const uint4*)(Arow1 + kb + 16);
    if (kk + 32 < Kk) {
      __builtin_prefetch(Arow0 + kb + 64, 0, 1);
      __builtin_prefetch(Arow1 + kb + 64, 0, 1);
    }
    int kb16 = kk + half * 8;                  // B frag: K 0..7 / 8..15 by lane-half, +16 upper
#pragma unroll
    for (int t = 0; t < 4; ++t) {
      Frag bfr;
      bfr.q[0] = *(const uint4*)(Wp[t] + kb16);
      bfr.q[1] = *(const uint4*)(Wp[t] + kb16 + 16);
      acc[0][t] = __builtin_amdgcn_wmma_f32_16x16x32_bf16(
          false, a0.v, false, bfr.v, (short)0, acc[0][t], false, false);
      acc[1][t] = __builtin_amdgcn_wmma_f32_16x16x32_bf16(
          false, a1.v, false, bfr.v, (short)0, acc[1][t], false, false);
    }
  }

#pragma unroll
  for (int t = 0; t < 4; ++t) {
    int col = colBase + t * 16 + lr;
    if (col >= Nn) continue;
    float bsv = bias ? bias[col] : 0.f;
#pragma unroll
    for (int r = 0; r < 2; ++r) {
#pragma unroll
      for (int v = 0; v < 8; ++v) {
        int rrow = row0 + r * 16 + half * 8 + v;   // D layout: M = vgpr + 8*(lane/16)
        float val = acc[r][t][v] + bsv;
        if (act == 1)      val = silu_f(val);
        else if (act == 2) val = (val > 20.f) ? val : log1pf(__expf(val));  // softplus
        size_t o = (size_t)rrow * ldc + col;
        if (mul) val *= mul[o];
        if (res) val += res[o];
        if (outF) outF[o] = val;
        if (outH) outH[o] = f2bf(val);
      }
    }
  }
}

// ---------------- depthwise causal conv (K=4) + SiLU ----------------
// xzh: BL x 4096 bf16 (x_in = cols 0..DI-1). Writes x_c fp32 + bf16.
__global__ __launch_bounds__(256) void conv_silu_kernel(const unsigned short* __restrict__ xzh,
                                                        const float* __restrict__ conv_w,
                                                        const float* __restrict__ conv_b,
                                                        float* __restrict__ xcF,
                                                        unsigned short* __restrict__ xcH) {
  size_t idx = (size_t)blockIdx.x * blockDim.x + threadIdx.x;   // over BL*DI
  int d = (int)(idx % DI);
  int l = (int)((idx / DI) % LL);
  int b = (int)(idx / ((size_t)DI * LL));
  float acc = conv_b[d];
#pragma unroll
  for (int k = 0; k < KC; ++k) {
    int ll = l - (KC - 1) + k;
    if (ll >= 0) {
      float xv = bf2f(xzh[((size_t)b * LL + ll) * N2 + d]);
      acc += conv_w[d * KC + k] * xv;
    }
  }
  acc = silu_f(acc);
  size_t o = ((size_t)b * LL + l) * DI + d;
  xcF[o] = acc;
  xcH[o] = f2bf(acc);
}

// ---------------- selective scan: one (b,d) per thread; B/C staged in LDS ----------------
// Fused epilogue: y = (scan + u*D[d]) * silu(z)  -> bf16 for out_proj GEMM.
__global__ __launch_bounds__(256) void scan_kernel(const float* __restrict__ dtf,
                                                   const float* __restrict__ xcF,
                                                   const float* __restrict__ xdbl,   // BL x 96
                                                   const unsigned short* __restrict__ xzh,
                                                   const float* __restrict__ A_log,
                                                   const float* __restrict__ Dvec,
                                                   unsigned short* __restrict__ ygh) {
  int b = blockIdx.x >> 3;                       // DI/256 = 8 blocks per batch
  int d = ((blockIdx.x & 7) << 8) + threadIdx.x;
  float Ar[NS];
#pragma unroll
  for (int n = 0; n < NS; ++n) Ar[n] = -__expf(A_log[d * NS + n]);
  float Dd = Dvec[d];
  float h[NS];
#pragma unroll
  for (int n = 0; n < NS; ++n) h[n] = 0.f;
  __shared__ float bc[2 * NS];

  for (int l = 0; l < LL; ++l) {
    size_t tok = (size_t)b * LL + l;
    if (threadIdx.x < 2 * NS) bc[threadIdx.x] = xdbl[tok * XD + DRr + threadIdx.x];
    __syncthreads();
    float dtv = dtf[tok * DI + d];
    float u   = xcF[tok * DI + d];
    float y = 0.f;
#pragma unroll
    for (int n = 0; n < NS; ++n) {
      float dA = __expf(dtv * Ar[n]);
      h[n] = dA * h[n] + (dtv * bc[n]) * u;
      y += h[n] * bc[NS + n];
    }
    float zf = bf2f(xzh[tok * N2 + DI + d]);
    float out = (y + u * Dd) * silu_f(zf);
    ygh[tok * DI + d] = f2bf(out);
    __syncthreads();
  }
}

// ---------------- host launcher ----------------
extern "C" void kernel_launch(void* const* d_in, const int* in_sizes, int n_in,
                              void* d_out, int out_size, void* d_ws, size_t ws_size,
                              hipStream_t stream) {
  (void)in_sizes; (void)n_in; (void)out_size; (void)ws_size;
  const float* x         = (const float*)d_in[0];
  const float* in_proj_w = (const float*)d_in[1];
  const float* conv_w    = (const float*)d_in[2];
  const float* conv_b    = (const float*)d_in[3];
  const float* x_proj_w  = (const float*)d_in[4];
  const float* dt_proj_w = (const float*)d_in[5];
  const float* dt_proj_b = (const float*)d_in[6];
  const float* A_log     = (const float*)d_in[7];
  const float* Dvec      = (const float*)d_in[8];
  const float* out_proj_w= (const float*)d_in[9];
  const float* hgd_w1    = (const float*)d_in[10];
  const float* hgd_b1    = (const float*)d_in[11];
  const float* hgd_w2    = (const float*)d_in[12];
  const float* hgd_b2    = (const float*)d_in[13];
  const float* hgf_wm    = (const float*)d_in[14];
  const float* hgf_bm    = (const float*)d_in[15];
  const float* hgf_wr    = (const float*)d_in[16];
  const float* hgf_br    = (const float*)d_in[17];
  const float* hgf_wf    = (const float*)d_in[18];
  const float* hgf_bf    = (const float*)d_in[19];
  float* out = (float*)d_out;

  // bump allocator over workspace
  char* p = (char*)d_ws;
  auto alloc = [&](size_t bytes) -> void* {
    void* r = (void*)p;
    p += (bytes + 255) & ~(size_t)255;
    return r;
  };
  typedef unsigned short bf16s;

  bf16s* w_inproj = (bf16s*)alloc((size_t)N2 * HH * 2);
  bf16s* w_xproj  = (bf16s*)alloc((size_t)XD * DI * 2);
  bf16s* w_dtproj = (bf16s*)alloc((size_t)DI * DRr * 2);
  bf16s* w_outprj = (bf16s*)alloc((size_t)HH * DI * 2);
  bf16s* w_hgd1   = (bf16s*)alloc((size_t)256 * HH * 2);
  bf16s* w_hgd2   = (bf16s*)alloc((size_t)HH * 256 * 2);
  bf16s* w_hgfm   = (bf16s*)alloc((size_t)256 * HH * 2);
  bf16s* w_hgfr   = (bf16s*)alloc((size_t)256 * HH * 2);
  bf16s* w_hgff   = (bf16s*)alloc((size_t)HH * 256 * 2);

  bf16s* x1h   = (bf16s*)alloc((size_t)BL * HH * 2);
  bf16s* hd1h  = (bf16s*)alloc((size_t)BL * 256 * 2);
  bf16s* hdh   = (bf16s*)alloc((size_t)BL * HH * 2);
  bf16s* xzh   = (bf16s*)alloc((size_t)BL * N2 * 2);
  float* xcF   = (float*)alloc((size_t)BL * DI * 4);
  bf16s* xcH   = (bf16s*)alloc((size_t)BL * DI * 2);
  float* xdbl  = (float*)alloc((size_t)BL * XD * 4);
  bf16s* xdblh = (bf16s*)alloc((size_t)BL * XD * 2);
  float* dtf   = (float*)alloc((size_t)BL * DI * 4);
  bf16s* ygh   = (bf16s*)alloc((size_t)BL * DI * 2);
  bf16s* x12h  = (bf16s*)alloc((size_t)BL * HH * 2);
  float* rbuf  = (float*)alloc((size_t)BL * 256 * 4);
  bf16s* mrh   = (bf16s*)alloc((size_t)BL * 256 * 2);

  auto cvt = [&](const float* src, bf16s* dst, long long n) {
    cvt_bf16_kernel<<<1024, 256, 0, stream>>>(src, dst, n);
  };
  cvt(in_proj_w, w_inproj, (long long)N2 * HH);
  cvt(x_proj_w,  w_xproj,  (long long)XD * DI);
  cvt(dt_proj_w, w_dtproj, (long long)DI * DRr);
  cvt(out_proj_w,w_outprj, (long long)HH * DI);
  cvt(hgd_w1,    w_hgd1,   (long long)256 * HH);
  cvt(hgd_w2,    w_hgd2,   (long long)HH * 256);
  cvt(hgf_wm,    w_hgfm,   (long long)256 * HH);
  cvt(hgf_wr,    w_hgfr,   (long long)256 * HH);
  cvt(hgf_wf,    w_hgff,   (long long)HH * 256);

  auto gemm = [&](const bf16s* A, int lda, const bf16s* W, int ldw,
                  const float* bias, const float* mul, const float* res,
                  float* oF, bf16s* oH, int M, int Nn, int Kk, int ldc, int act) {
    dim3 g((Nn + 127) / 128, M / 128);
    gemm_bf16_wmma<<<g, 256, 0, stream>>>(A, lda, W, ldw, bias, mul, res,
                                          oF, oH, M, Nn, Kk, ldc, act);
  };

  // 1) x1 = layernorm(x)
  ln_kernel<<<BL, 256, 0, stream>>>(x, x1h);
  // 2) hd1 = silu(x1 @ hgd_w1^T + b1)
  gemm(x1h, HH, w_hgd1, HH, hgd_b1, nullptr, nullptr, nullptr, hd1h, BL, 256, HH, 256, 1);
  // 3) hd = hd1 @ hgd_w2^T + b2
  gemm(hd1h, 256, w_hgd2, 256, hgd_b2, nullptr, nullptr, nullptr, hdh, BL, HH, 256, HH, 0);
  // 4) xz = hd @ in_proj_w^T   (x_in | z)
  gemm(hdh, HH, w_inproj, HH, nullptr, nullptr, nullptr, nullptr, xzh, BL, N2, HH, N2, 0);
  // 5) x_c = silu(causal_conv(x_in))
  conv_silu_kernel<<<(BL * DI) / 256, 256, 0, stream>>>(xzh, conv_w, conv_b, xcF, xcH);
  // 6) x_dbl = x_c @ x_proj_w^T   (dt_lo | B | C), N=96
  gemm(xcH, DI, w_xproj, DI, nullptr, nullptr, nullptr, xdbl, xdblh, BL, XD, DI, XD, 0);
  // 7) dt = softplus(dt_lo @ dt_proj_w^T + dt_proj_b)
  gemm(xdblh, XD, w_dtproj, DRr, dt_proj_b, nullptr, nullptr, dtf, nullptr, BL, DI, DRr, DI, 2);
  // 8) selective scan + D skip + silu(z) gate -> ygh
  scan_kernel<<<BB * (DI / 256), 256, 0, stream>>>(dtf, xcF, xdbl, xzh, A_log, Dvec, ygh);
  // 9) x1_2 = y @ out_proj_w^T
  gemm(ygh, DI, w_outprj, DI, nullptr, nullptr, nullptr, nullptr, x12h, BL, HH, DI, HH, 0);
  // 10) r = silu(x1 @ hgf_wr^T + br)
  gemm(x1h, HH, w_hgfr, HH, hgf_br, nullptr, nullptr, rbuf, nullptr, BL, 256, HH, 256, 1);
  // 11) mr = silu(x1_2 @ hgf_wm^T + bm) * r
  gemm(x12h, HH, w_hgfm, HH, hgf_bm, rbuf, nullptr, nullptr, mrh, BL, 256, HH, 256, 1);
  // 12) out = x + (mr @ hgf_wf^T + bf)
  gemm(mrh, 256, w_hgff, 256, hgf_bf, nullptr, x, out, nullptr, BL, HH, 256, HH, 0);
}
